// GraphAttentionLayer_78245714199066
// MI455X (gfx1250) — matrix operand order repaired and checked by reading, not compiled, not run on hardware
//
#include <hip/hip_runtime.h>
#include <hip/hip_bf16.h>
#include <hip/hip_fp16.h>

#define NN 8192
#define FF 512
#define LALPHA 0.2f
#define NEG_BIG (-9000000000000000.0f)

typedef _Float16 f16;
typedef __attribute__((ext_vector_type(16))) _Float16     v16h;
typedef __attribute__((ext_vector_type(8)))  _Float16     v8h;
typedef __attribute__((ext_vector_type(8)))  float        v8f;
typedef __attribute__((ext_vector_type(4)))  float        v4f;
typedef __attribute__((ext_vector_type(4)))  int          v4i;
typedef __attribute__((ext_vector_type(8)))  int          v8i;
typedef __attribute__((ext_vector_type(4)))  unsigned int v4u;

#if defined(__has_builtin)
#if __has_builtin(__builtin_amdgcn_tensor_load_to_lds)
#define HAVE_TDM 1
#else
#define HAVE_TDM 0
#endif
#else
#define HAVE_TDM 0
#endif

#if __has_include(<hip/amd_detail/amd_gfx1250_TDM.h>)
#define TDM_SIX_ARG 1
#else
#define TDM_SIX_ARG 0
#endif

// flat pointer to shared memory -> LDS byte address (flat LDS aperture keeps the
// wave-relative LDS address in the low 32 bits)
__device__ __forceinline__ unsigned lds_off(const void* p) {
    return (unsigned)(unsigned long long)(uintptr_t)p;
}

#if HAVE_TDM
// Issue a 2-D TDM load: tile (tile_d0 x tile_d1) of 2-byte elements from a row-major
// tensor with row stride stride0 (elements) into LDS at lds_byte (packed, 2*tile_d0 per row).
__device__ __forceinline__ void tdm_load_2d_b16(unsigned lds_byte, const void* gaddr,
                                                unsigned tensor_d0, unsigned tensor_d1,
                                                unsigned tile_d0, unsigned tile_d1,
                                                unsigned long long stride0) {
    unsigned long long ga = (unsigned long long)(uintptr_t)gaddr;
    v4u g0;
    g0.x = 1u;                                               // count=1, user mode, no gather
    g0.y = lds_byte;                                         // lds_addr
    g0.z = (unsigned)(ga & 0xFFFFFFFFu);                     // global_addr[31:0]
    g0.w = (unsigned)((ga >> 32) & 0x1FFFFFFu) | (2u << 30); // global_addr[56:32] | type=2
    v8i g1;
    g1[0] = (1 << 16);                                       // data_size=2B, no barrier/iter/pad
    g1[1] = (int)((tensor_d0 & 0xFFFFu) << 16);              // tensor_dim0[15:0]
    g1[2] = (int)(((tensor_d0 >> 16) & 0xFFFFu) | ((tensor_d1 & 0xFFFFu) << 16));
    g1[3] = (int)(((tensor_d1 >> 16) & 0xFFFFu) | ((tile_d0 & 0xFFFFu) << 16));
    g1[4] = (int)(tile_d1 & 0xFFFFu);                        // tile_dim1, tile_dim2=0 (2D)
    g1[5] = (int)(unsigned)(stride0 & 0xFFFFFFFFu);          // tensor_dim0_stride[31:0]
    g1[6] = (int)(unsigned)((stride0 >> 32) & 0xFFFFu);      // stride[47:32], dim1_stride=0
    g1[7] = 0;
    v4i z4 = {0, 0, 0, 0};
#if TDM_SIX_ARG
    v8i z8 = {0, 0, 0, 0, 0, 0, 0, 0};
    __builtin_amdgcn_tensor_load_to_lds(g0, g1, z4, z4, z8, 0);
#else
    __builtin_amdgcn_tensor_load_to_lds(g0, g1, z4, z4, 0);
#endif
}
#endif

// ---------------- conversion kernels ----------------
__global__ __launch_bounds__(256) void k_cvtX(const float* __restrict__ X, f16* __restrict__ Xh) {
    int idx = blockIdx.x * 256 + threadIdx.x;          // NN*FF elements
    Xh[idx] = (f16)X[idx];
}

__global__ __launch_bounds__(256) void k_cvtW(const float* __restrict__ W, f16* __restrict__ WhT) {
    int idx = blockIdx.x * 256 + threadIdx.x;          // FF*FF elements, idx = n*FF + k
    int n = idx >> 9;
    int k = idx & (FF - 1);
    WhT[idx] = (f16)W[k * FF + n];                     // transpose: WhT[n][k] = W[k][n]
}

// ---------------- h = X @ W  (f16 WMMA, f32 accum) ----------------
__global__ __launch_bounds__(256) void k_gemm_h(const f16* __restrict__ Xh, const f16* __restrict__ WhT,
                                                float* __restrict__ h, f16* __restrict__ hhT) {
    const int lane = threadIdx.x & 31;
    const int wid  = blockIdx.x * 8 + (threadIdx.x >> 5);
    const int rt   = wid >> 3;                         // row tile (16 rows)
    const int ct   = wid & 7;                          // col tile (64 cols)
    const int r    = lane & 15;
    const int half = lane >> 4;

    v8f c[4] = {};
    const f16* aptr = Xh + (size_t)(rt * 16 + r) * FF + half * 16;

    for (int k0 = 0; k0 < FF; k0 += 32) {
        v16h a;
        *((v8h*)&a)       = *(const v8h*)(aptr + k0);
        *(((v8h*)&a) + 1) = *(const v8h*)(aptr + k0 + 8);
#pragma unroll
        for (int t = 0; t < 4; ++t) {
            const int col = ct * 64 + t * 16 + r;
            v16h b = *(const v16h*)(WhT + (size_t)col * FF + k0 + half * 16);
            c[t] = __builtin_amdgcn_wmma_f32_16x16x32_f16(false, a, false, b, (short)0, c[t], false, false);
        }
    }
#pragma unroll
    for (int t = 0; t < 4; ++t) {
#pragma unroll
        for (int v = 0; v < 8; ++v) {
            const int row = rt * 16 + half * 8 + v;
            const int col = ct * 64 + t * 16 + r;
            const float val = c[t][v];
            h[(size_t)row * FF + col] = val;
            hhT[(size_t)col * NN + row] = (f16)val;
        }
    }
}

// ---------------- s1 = h@a1, s2 = h@a2 (one wave per row) ----------------
__global__ __launch_bounds__(256) void k_scores(const float* __restrict__ h, const float* __restrict__ attn,
                                                float* __restrict__ s1, float* __restrict__ s2) {
    const int lane = threadIdx.x & 31;
    const int row  = blockIdx.x * 8 + (threadIdx.x >> 5);
    const float* hp = h + (size_t)row * FF;
    float a1 = 0.f, a2 = 0.f;
    for (int t = 0; t < FF; t += 32) {
        float x = hp[t + lane];
        a1 += x * attn[t + lane];
        a2 += x * attn[FF + t + lane];
    }
    for (int off = 16; off; off >>= 1) {
        a1 += __shfl_xor(a1, off, 32);
        a2 += __shfl_xor(a2, off, 32);
    }
    if (lane == 0) { s1[row] = a1; s2[row] = a2; }
}

// ---------------- pass 1: per-row softmax stats, b128 adjacency streaming ----------------
__global__ __launch_bounds__(256) void k_rowstats(const int* __restrict__ adj, const float* __restrict__ s1,
                                                  const float* __restrict__ s2,
                                                  float* __restrict__ rowmax, float* __restrict__ rowrz) {
    const int lane = threadIdx.x & 31;
    const int row  = blockIdx.x * 8 + (threadIdx.x >> 5);
    const float si = s1[row];
    const int* ap  = adj + (size_t)row * NN;
    float m = -INFINITY, z = 0.f;
    for (int t = 0; t < NN; t += 128) {
        const int jb = t + lane * 4;
        v4i av = *(const v4i*)(ap + jb);
        v4f sv = *(const v4f*)(s2 + jb);
        __builtin_prefetch(ap + jb + 1024, 0, 1);
#pragma unroll
        for (int u = 0; u < 4; ++u) {
            float e0 = si + sv[u];
            float e  = (av[u] > 0) ? (e0 > 0.f ? e0 : LALPHA * e0) : NEG_BIG;
            if (e > m) { z = z * __expf(m - e) + 1.f; m = e; }
            else       { z += __expf(e - m); }
        }
    }
    for (int off = 16; off; off >>= 1) {
        float mo = __shfl_xor(m, off, 32);
        float zo = __shfl_xor(z, off, 32);
        float M  = fmaxf(m, mo);
        z = z * __expf(m - M) + zo * __expf(mo - M);
        m = M;
    }
    if (lane == 0) { rowmax[row] = m; rowrz[row] = 1.f / z; }
}

// ---------------- pass 2: fused P = exp(e - m), h' = (P @ h) / Z, elu ----------------
// block: 32 rows x 512 cols, 8 waves; wave tile 16 rows x 128 cols (8 accumulators).
// B tile (32x512 f16) double-buffered in LDS, staged by the Tensor Data Mover;
// P tile (32x32 f16) computed by VALU into LDS each step.
__global__ __launch_bounds__(256) void k_attn_out(const int* __restrict__ adj, const f16* __restrict__ hhT,
                                                  const float* __restrict__ s1, const float* __restrict__ s2,
                                                  const float* __restrict__ rowmax, const float* __restrict__ rowrz,
                                                  float* __restrict__ out) {
    __shared__ f16 Bl[2][512 * 32];   // [buf][col][k]  2 x 32 KB
    __shared__ f16 PA[32 * 32];       // [rowloc][k]    2 KB

    const int tid     = threadIdx.x;
    const int lane    = tid & 31;
    const int w       = tid >> 5;
    const int rowbase = blockIdx.x * 32;
    const int NSTEP   = NN / 32;

    // P-tile producer role: 32 rows x 32 k by 256 threads -> 4 entries each
    const int parow = tid >> 3;
    const int pak   = (tid & 7) * 4;
    const float si  = s1[rowbase + parow];
    const float mi  = rowmax[rowbase + parow];
    const int* aprow = adj + (size_t)(rowbase + parow) * NN + pak;

    // wave tile coordinates
    const int wrow = (w & 1) * 16;       // local row offset 0/16
    const int colq = (w >> 1) * 128;     // col quarter
    const int r    = lane & 15;
    const int half = lane >> 4;

    v8f c[8] = {};

#if HAVE_TDM
    // prologue: DMA first B tile into buffer 0
    if (w == 0) {
        tdm_load_2d_b16(lds_off(&Bl[0][0]), hhT, NN, FF, 32, FF, (unsigned long long)NN);
    }
#endif

    for (int s = 0; s < NSTEP; ++s) {
        const int j0  = s * 32;
        const int buf = s & 1;
        __syncthreads();   // previous tiles fully consumed (Bl[buf^1] of step s-1, PA)

#if HAVE_TDM
        // prefetch next B tile into the other buffer while this step computes
        if (w == 0 && s + 1 < NSTEP) {
            tdm_load_2d_b16(lds_off(&Bl[buf ^ 1][0]), hhT + (j0 + 32), NN, FF, 32, FF,
                            (unsigned long long)NN);
        }
#else
        // fallback: cooperative copy of the current B tile
#pragma unroll
        for (int cc = 0; cc < 2; ++cc) {
            const int col = tid + cc * 256;
            const v8h* src = (const v8h*)(hhT + (size_t)col * NN + j0);
            v8h* dst = (v8h*)(&Bl[buf][col * 32]);
            dst[0] = src[0]; dst[1] = src[1]; dst[2] = src[2]; dst[3] = src[3];
        }
#endif
        // stage P: 4 entries per thread (adjacency mask + leaky_relu + exp)
        {
            v4i av = *(const v4i*)(aprow + j0);
            __builtin_prefetch(aprow + j0 + 256, 0, 1);
            f16* pa = PA + parow * 32 + pak;
#pragma unroll
            for (int u = 0; u < 4; ++u) {
                float e0 = si + s2[j0 + pak + u];
                float e  = (av[u] > 0) ? (e0 > 0.f ? e0 : LALPHA * e0) : NEG_BIG;
                pa[u] = (f16)__expf(e - mi);     // masked -> exp(very negative) == 0
            }
        }
#if HAVE_TDM
        if (w == 0) {
            if (s + 1 < NSTEP) __builtin_amdgcn_s_wait_tensorcnt(1);  // step-s tile done (in-order)
            else               __builtin_amdgcn_s_wait_tensorcnt(0);
        }
#endif
        __syncthreads();

        // A operand from PA (two contiguous 16B runs per lane)
        v16h a;
        {
            const f16* par = PA + (wrow + r) * 32 + half * 16;
            *((v8h*)&a)       = *(const v8h*)par;
            *(((v8h*)&a) + 1) = *(const v8h*)(par + 8);
        }
#pragma unroll
        for (int t = 0; t < 8; ++t) {
            const int coll = colq + t * 16 + r;
            v16h b = *(const v16h*)(&Bl[buf][coll * 32 + half * 16]);
            c[t] = __builtin_amdgcn_wmma_f32_16x16x32_f16(false, a, false, b, (short)0, c[t], false, false);
        }
    }

    // epilogue: scale by 1/Z, elu, store
#pragma unroll
    for (int v = 0; v < 8; ++v) {
        const int row = rowbase + wrow + half * 8 + v;
        const float rz = rowrz[row];
#pragma unroll
        for (int t = 0; t < 8; ++t) {
            const int col = colq + t * 16 + r;
            float val = c[t][v] * rz;
            out[(size_t)row * FF + col] = (val > 0.f) ? val : (__expf(val) - 1.f);
        }
    }
}

// ---------------- launcher ----------------
extern "C" void kernel_launch(void* const* d_in, const int* in_sizes, int n_in,
                              void* d_out, int out_size, void* d_ws, size_t ws_size,
                              hipStream_t stream) {
    const int*   adj  = (const int*)d_in[0];
    const float* X    = (const float*)d_in[1];
    const float* W    = (const float*)d_in[2];
    const float* attn = (const float*)d_in[3];
    float* out = (float*)d_out;

    char* ws = (char*)d_ws;
    f16*   Xh     = (f16*)(ws);                                            // 8 MB
    f16*   WhT    = (f16*)(ws + ((size_t)8 << 20));                        // 512 KB
    f16*   hhT    = (f16*)(ws + ((size_t)8 << 20) + ((size_t)512 << 10));  // 8 MB
    float* s1     = (float*)(ws + ((size_t)16 << 20) + ((size_t)512 << 10));
    float* s2     = s1 + NN;
    float* rowmax = s2 + NN;
    float* rowrz  = rowmax + NN;
    float* h      = out;   // park fp32 h in d_out; final kernel overwrites every element

    k_cvtX<<<(NN * FF) / 256, 256, 0, stream>>>(X, Xh);
    k_cvtW<<<(FF * FF) / 256, 256, 0, stream>>>(W, WhT);
    k_gemm_h<<<NN / 16, 256, 0, stream>>>(Xh, WhT, h, hhT);   // 512 blocks x 8 waves
    k_scores<<<NN / 8, 256, 0, stream>>>(h, attn, s1, s2);
    k_rowstats<<<NN / 8, 256, 0, stream>>>(adj, s1, s2, rowmax, rowrz);
    k_attn_out<<<NN / 32, 256, 0, stream>>>(adj, hhT, s1, s2, rowmax, rowrz, out);
}
